// GobalSuperPixelAttention_38311108280863
// MI455X (gfx1250) — matrix operand order repaired and checked by reading, not compiled, not run on hardware
//
#include <hip/hip_runtime.h>
#include <math.h>

// ---------------- problem constants ----------------
#define Bn 2
#define Cc 192
#define Hh 192
#define Wd 192
#define HHs 24
#define WWs 24
#define Mm 576            // HHs*WWs superpixels
#define Pp 64             // pixels per superpixel
#define Nn 36864          // Hh*Wd
#define HD 48             // head dim
#define NH 4
#define TC 576            // 3*C

// ---------------- WMMA helpers (bf16 16x16x32, f32 accum) ----------------
typedef __attribute__((ext_vector_type(16))) __bf16 bfrag;
typedef __attribute__((ext_vector_type(8)))  float v8f;

#define WMMA_BF16(c, a, b) \
  (c) = __builtin_amdgcn_wmma_f32_16x16x32_bf16(false, (a), false, (b), (short)0, (c), false, false)

__device__ __forceinline__ v8f v8f_zero() {
  v8f z;
#pragma unroll
  for (int i = 0; i < 8; ++i) z[i] = 0.f;
  return z;
}

// A-frag from row-major (rows x K) tile origin; per ISA layout:
// lane<16: elems 0..7 -> K 0..7 ; 8..15 -> K 16..23 ; lane>=16 shifted by 8.
__device__ __forceinline__ bfrag load_a_rm(const __bf16* A, int lda, int lane) {
  int row = lane & 15, kb = (lane >> 4) << 3;
  const __bf16* r = A + (size_t)row * lda + kb;
  bfrag f;
#pragma unroll
  for (int j = 0; j < 8; ++j) { f[j] = r[j]; f[8 + j] = r[16 + j]; }
  return f;
}
// A-frag where logical A is the transpose of a row-major buffer (buffer is (K x rows))
__device__ __forceinline__ bfrag load_a_cm(const __bf16* At, int lda, int lane) {
  int row = lane & 15, kb = (lane >> 4) << 3;
  bfrag f;
#pragma unroll
  for (int j = 0; j < 8; ++j) {
    f[j]     = At[(size_t)(kb + j) * lda + row];
    f[8 + j] = At[(size_t)(16 + kb + j) * lda + row];
  }
  return f;
}
// B-frag from row-major (N x K) buffer holding B^T: b[j] = B[k][col] = Bt[col][k]
__device__ __forceinline__ bfrag load_b_cm(const __bf16* Bt, int ldb, int lane) {
  int col = lane & 15, kb = (lane >> 4) << 3;
  const __bf16* r = Bt + (size_t)col * ldb + kb;
  bfrag f;
#pragma unroll
  for (int j = 0; j < 16; ++j) f[j] = r[j + (j & 8)];
  return f;
}
// B-frag from row-major (K x N) buffer
__device__ __forceinline__ bfrag load_b_rm(const __bf16* Bm, int ldb, int lane) {
  int col = lane & 15, kb = (lane >> 4) << 3;
  bfrag f;
#pragma unroll
  for (int j = 0; j < 16; ++j) f[j] = Bm[(size_t)(kb + j + (j & 8)) * ldb + col];
  return f;
}
// guarded versions for K=48 (pad to 64 with zeros)
__device__ __forceinline__ bfrag load_a_rm_g(const __bf16* A, int lda, int lane, int kc, int klim) {
  int row = lane & 15, kb = (lane >> 4) << 3;
  const __bf16* r = A + (size_t)row * lda;
  bfrag f;
#pragma unroll
  for (int j = 0; j < 8; ++j) {
    int k0 = kc + kb + j, k1 = kc + 16 + kb + j;
    f[j]     = (k0 < klim) ? r[k0] : (__bf16)0.f;
    f[8 + j] = (k1 < klim) ? r[k1] : (__bf16)0.f;
  }
  return f;
}
__device__ __forceinline__ bfrag load_b_cm_g(const __bf16* Bt, int ldb, int lane, int kc, int klim) {
  int col = lane & 15, kb = (lane >> 4) << 3;
  const __bf16* r = Bt + (size_t)col * ldb;
  bfrag f;
#pragma unroll
  for (int j = 0; j < 16; ++j) { int k = kc + kb + j + (j & 8); f[j] = (k < klim) ? r[k] : (__bf16)0.f; }
  return f;
}

// ================= kernels =================

// pack x_ref into (B,N,C) bf16 transposed pixels + fp32 squared norms
__global__ __launch_bounds__(256, 1)
void k_pack_xref(const float* __restrict__ xr, __bf16* __restrict__ XrT,
                 float* __restrict__ pnorm) {
  __shared__ float tile[64][Cc + 1];
  int blk = blockIdx.x, b = blk / (Nn / 64), n0 = (blk % (Nn / 64)) * 64;
  const float* src = xr + (size_t)b * Cc * Nn;
  for (int idx = threadIdx.x; idx < 64 * Cc; idx += 256) {
    int c = idx >> 6, j = idx & 63;
    tile[j][c] = src[(size_t)c * Nn + n0 + j];
  }
  __syncthreads();
  __bf16* dst = XrT + ((size_t)b * Nn + n0) * Cc;
  for (int idx = threadIdx.x; idx < 64 * Cc; idx += 256) {
    int j = idx / Cc, c = idx % Cc;
    dst[(size_t)j * Cc + c] = (__bf16)tile[j][c];
  }
  if (threadIdx.x < 64) {
    float s = 0.f;
    for (int c = 0; c < Cc; ++c) { float v = tile[threadIdx.x][c]; s += v * v; }
    pnorm[(size_t)b * Nn + n0 + threadIdx.x] = s;
  }
}

// block-average centroids -> centT bf16 (B,M,C) + cnorm fp32 (B,M)
__global__ __launch_bounds__(256, 1)
void k_centroid(const float* __restrict__ xr, __bf16* __restrict__ centT,
                float* __restrict__ cnorm) {
  __shared__ float tile[Pp][Cc + 1];
  __shared__ float cn;
  int blk = blockIdx.x, b = blk / Mm, m = blk % Mm;
  int sy = m / WWs, sx = m % WWs;
  const float* src = xr + (size_t)b * Cc * Nn;
  for (int idx = threadIdx.x; idx < Pp * Cc; idx += 256) {
    int c = idx >> 6, p = idx & 63;
    tile[p][c] = src[((size_t)c * Hh + sy * 8 + (p >> 3)) * Wd + sx * 8 + (p & 7)];
  }
  if (threadIdx.x == 0) cn = 0.f;
  __syncthreads();
  if (threadIdx.x < Cc) {
    float s = 0.f;
    for (int p = 0; p < Pp; ++p) s += tile[p][threadIdx.x];
    float mean = s * (1.0f / 64.0f);
    centT[((size_t)b * Mm + m) * Cc + threadIdx.x] = (__bf16)mean;
    atomicAdd(&cn, mean * mean);
  }
  __syncthreads();
  if (threadIdx.x == 0) cnorm[(size_t)b * Mm + m] = cn;
}

// generic f32 -> bf16 pack
__global__ __launch_bounds__(256, 1)
void k_pack_w(const float* __restrict__ s, __bf16* __restrict__ d, int n) {
  int i = blockIdx.x * 256 + threadIdx.x;
  if (i < n) d[i] = (__bf16)s[i];
}

// soft kmeans assignment: logits = -sqrt(max(|p|^2+|c|^2-2 p.c,0)); softmax over M.
// 32 pixels/block, 8 waves = 2 row-tiles x 4 col-quarters (144 cols -> 9 tiles/wave).
__global__ __launch_bounds__(256, 1)
void k_assign(const __bf16* __restrict__ XrT, const __bf16* __restrict__ centT,
              const float* __restrict__ pnorm, const float* __restrict__ cnorm,
              __bf16* __restrict__ Abuf, float* __restrict__ sw) {
  __shared__ __bf16 XL[32][Cc + 8];
  __shared__ float pnL[32];
  __shared__ float rmax4[4][32];
  __shared__ float rsum4[4][32];
  int blk = blockIdx.x, b = blk / (Nn / 32), n0 = (blk % (Nn / 32)) * 32;
  int tid = threadIdx.x, lane = tid & 31, w = tid >> 5;
  const __bf16* xs = XrT + ((size_t)b * Nn + n0) * Cc;
  for (int idx = tid; idx < 32 * Cc; idx += 256) {
    int r = idx / Cc, c = idx % Cc;
    XL[r][c] = xs[(size_t)r * Cc + c];
  }
  if (tid < 32) pnL[tid] = pnorm[(size_t)b * Nn + n0 + tid];
  __syncthreads();
  int rt = w & 1, cq = w >> 1, colbase = cq * 144;
  int lcol = lane & 15, roff = (lane < 16) ? 0 : 8;
  int rowb = rt * 16 + roff;
  v8f acc[9];
#pragma unroll
  for (int t = 0; t < 9; ++t) acc[t] = v8f_zero();
  const __bf16* cb = centT + (size_t)b * Mm * Cc;
  for (int kc = 0; kc < Cc; kc += 32) {
    bfrag a = load_a_rm(&XL[rt * 16][kc], Cc + 8, lane);
#pragma unroll
    for (int t = 0; t < 9; ++t) {
      bfrag bb = load_b_cm(cb + (size_t)(colbase + t * 16) * Cc + kc, Cc, lane);
      WMMA_BF16(acc[t], a, bb);
    }
  }
#pragma unroll
  for (int t = 0; t < 9; ++t) {
    float cnv = cnorm[(size_t)b * Mm + colbase + t * 16 + lcol];
#pragma unroll
    for (int i = 0; i < 8; ++i) {
      float d2 = pnL[rowb + i] + cnv - 2.f * acc[t][i];
      acc[t][i] = -sqrtf(fmaxf(d2, 0.f));
    }
  }
  float mrow[8], srow[8];
#pragma unroll
  for (int i = 0; i < 8; ++i) {
    float mx = -1e30f;
#pragma unroll
    for (int t = 0; t < 9; ++t) mx = fmaxf(mx, acc[t][i]);
#pragma unroll
    for (int o = 1; o < 16; o <<= 1) mx = fmaxf(mx, __shfl_xor(mx, o, 16));
    mrow[i] = mx;
  }
  if (lcol == 0)
#pragma unroll
    for (int i = 0; i < 8; ++i) rmax4[cq][rowb + i] = mrow[i];
  __syncthreads();
#pragma unroll
  for (int i = 0; i < 8; ++i) {
    int r = rowb + i;
    mrow[i] = fmaxf(fmaxf(rmax4[0][r], rmax4[1][r]), fmaxf(rmax4[2][r], rmax4[3][r]));
  }
#pragma unroll
  for (int i = 0; i < 8; ++i) {
    float s = 0.f;
#pragma unroll
    for (int t = 0; t < 9; ++t) s += __expf(acc[t][i] - mrow[i]);
#pragma unroll
    for (int o = 1; o < 16; o <<= 1) s += __shfl_xor(s, o, 16);
    srow[i] = s;
  }
  if (lcol == 0)
#pragma unroll
    for (int i = 0; i < 8; ++i) rsum4[cq][rowb + i] = srow[i];
  __syncthreads();
#pragma unroll
  for (int i = 0; i < 8; ++i) {
    int r = rowb + i;
    srow[i] = 1.f / (rsum4[0][r] + rsum4[1][r] + rsum4[2][r] + rsum4[3][r]);
  }
#pragma unroll
  for (int t = 0; t < 9; ++t) {
    int col = colbase + t * 16 + lcol;
    float cs = 0.f;
#pragma unroll
    for (int i = 0; i < 8; ++i) {
      float av = __expf(acc[t][i] - mrow[i]) * srow[i];
      cs += av;
      Abuf[((size_t)b * Nn + n0 + rowb + i) * Mm + col] = (__bf16)av;
    }
    atomicAdd(&sw[(size_t)b * Mm + col], cs);
  }
}

// wp = A^T X (M x C, K = N), split-K=16 with fp32 atomics
__global__ __launch_bounds__(256, 1)
void k_wp(const __bf16* __restrict__ Abuf, const __bf16* __restrict__ XrT,
          float* __restrict__ wp) {
  __shared__ __bf16 AT[64][34];
  __shared__ __bf16 BX[32][Cc + 8];
  int blk = blockIdx.x;
  int ks = blk % 16, mb = (blk / 16) % 9, b = blk / (16 * 9);
  int m0 = mb * 64;
  int tid = threadIdx.x, lane = tid & 31, w = tid >> 5;
  int rt = w & 3, cbase = (w >> 2) * 96;
  int roff = (lane < 16) ? 0 : 8, lcol = lane & 15;
  v8f acc[6];
#pragma unroll
  for (int t = 0; t < 6; ++t) acc[t] = v8f_zero();
  size_t k0 = (size_t)ks * (Nn / 16), k1 = k0 + Nn / 16;
  for (size_t kc = k0; kc < k1; kc += 32) {
    for (int idx = tid; idx < 32 * 64; idx += 256) {
      int r = idx >> 6, mj = idx & 63;
      AT[mj][r] = Abuf[((size_t)b * Nn + kc + r) * Mm + m0 + mj];
    }
    for (int idx = tid; idx < 32 * Cc; idx += 256) {
      int r = idx / Cc, c = idx % Cc;
      BX[r][c] = XrT[((size_t)b * Nn + kc + r) * Cc + c];
    }
    __syncthreads();
    bfrag a = load_a_rm(&AT[rt * 16][0], 34, lane);
#pragma unroll
    for (int t = 0; t < 6; ++t) {
      bfrag bb = load_b_rm(&BX[0][cbase + t * 16], Cc + 8, lane);
      WMMA_BF16(acc[t], a, bb);
    }
    __syncthreads();
  }
#pragma unroll
  for (int t = 0; t < 6; ++t)
#pragma unroll
    for (int i = 0; i < 8; ++i)
      atomicAdd(&wp[((size_t)b * Mm + m0 + rt * 16 + roff + i) * Cc + cbase + t * 16 + lcol],
                acc[t][i]);
}

__global__ __launch_bounds__(256, 1)
void k_sf_fin(const float* __restrict__ wp, const float* __restrict__ sw,
              __bf16* __restrict__ sfb) {
  size_t i = (size_t)blockIdx.x * 256 + threadIdx.x;
  if (i >= (size_t)Bn * Mm * Cc) return;
  sfb[i] = (__bf16)(wp[i] / (sw[i / Cc] + 1e-6f));
}

// per-superpixel: aff = softmax_9(pf . sfu * C^-0.5); scatter sums + features into centroids
__global__ __launch_bounds__(128, 1)
void k_aff(const float* __restrict__ xc, const __bf16* __restrict__ sfb,
           float* __restrict__ affG, float* __restrict__ sf2acc,
           float* __restrict__ asum) {
  __shared__ __bf16 pfL[64][Cc + 8];
  __shared__ __bf16 sfuL[16][Cc + 8];
  __shared__ __bf16 affL[64][16];
  __shared__ float wsumL[16];
  int blk = blockIdx.x, b = blk / Mm, m = blk % Mm;
  int sy = m / WWs, sx = m % WWs;
  int tid = threadIdx.x, lane = tid & 31, w = tid >> 5;
  const float* src = xc + (size_t)b * Cc * Nn;
  for (int idx = tid; idx < 64 * Cc; idx += 128) {
    int c = idx >> 6, p = idx & 63;
    pfL[p][c] = (__bf16)src[((size_t)c * Hh + sy * 8 + (p >> 3)) * Wd + sx * 8 + (p & 7)];
  }
  for (int idx = tid; idx < 16 * Cc; idx += 128) {
    int k = idx / Cc, c = idx % Cc;
    __bf16 v = (__bf16)0.f;
    if (k < 9) {
      int ny = sy + k / 3 - 1, nx = sx + k % 3 - 1;
      if (ny >= 0 && ny < HHs && nx >= 0 && nx < WWs)
        v = sfb[((size_t)b * Mm + ny * WWs + nx) * Cc + c];
    }
    sfuL[k][c] = v;
  }
  if (tid < 16) wsumL[tid] = 0.f;
  __syncthreads();
  int rt = w, lcol = lane & 15, roff = (lane < 16) ? 0 : 8;
  v8f s = v8f_zero();
  for (int kc = 0; kc < Cc; kc += 32) {
    bfrag a = load_a_rm(&pfL[rt * 16][kc], Cc + 8, lane);
    bfrag bb = load_b_cm(&sfuL[0][kc], Cc + 8, lane);
    WMMA_BF16(s, a, bb);
  }
  const float scale = rsqrtf((float)Cc);
  float csum = 0.f;
#pragma unroll
  for (int i = 0; i < 8; ++i) {
    float l = (lcol <= 8) ? s[i] * scale : -1e30f;
    float mx = l;
#pragma unroll
    for (int o = 1; o < 16; o <<= 1) mx = fmaxf(mx, __shfl_xor(mx, o, 16));
    float e = (lcol <= 8) ? __expf(l - mx) : 0.f;
    float ssum = e;
#pragma unroll
    for (int o = 1; o < 16; o <<= 1) ssum += __shfl_xor(ssum, o, 16);
    float av = e / ssum;
    int p = rt * 16 + roff + i;
    affL[p][lcol] = (__bf16)av;
    if (lcol <= 8) affG[(((size_t)b * Mm + m) * 64 + p) * 9 + lcol] = av;
    csum += av;
  }
  atomicAdd(&wsumL[lcol], csum);
  __syncthreads();
#pragma unroll
  for (int rr = 0; rr < 3; ++rr) {
    int ct = w * 3 + rr;
    v8f g = v8f_zero();
    for (int kc = 0; kc < 64; kc += 32) {
      bfrag a = load_a_cm(&pfL[kc][ct * 16], Cc + 8, lane);
      bfrag bb = load_b_rm(&affL[kc][0], 16, lane);
      WMMA_BF16(g, a, bb);
    }
    if (lcol <= 8) {
      int ny = sy + lcol / 3 - 1, nx = sx + lcol % 3 - 1;
      if (ny >= 0 && ny < HHs && nx >= 0 && nx < WWs) {
        float* dst = sf2acc + ((size_t)b * Mm + ny * WWs + nx) * Cc;
#pragma unroll
        for (int i = 0; i < 8; ++i) atomicAdd(&dst[ct * 16 + roff + i], g[i]);
      }
    }
  }
  if (tid < 9) {
    int ny = sy + tid / 3 - 1, nx = sx + tid % 3 - 1;
    if (ny >= 0 && ny < HHs && nx >= 0 && nx < WWs)
      atomicAdd(&asum[(size_t)b * Mm + ny * WWs + nx], wsumL[tid]);
  }
}

__global__ __launch_bounds__(256, 1)
void k_sf2_fin(const float* __restrict__ sf2acc, const float* __restrict__ asum,
               __bf16* __restrict__ sf2b) {
  size_t i = (size_t)blockIdx.x * 256 + threadIdx.x;
  if (i >= (size_t)Bn * Mm * Cc) return;
  sf2b[i] = (__bf16)(sf2acc[i] / (asum[i / Cc] + 1e-12f));
}

// qkvT[n][o] = sum_c sf2[n][c] W[o][c] + bias[o]
// 32 rows/block, 8 waves = 2 row-tiles x 4 col-quarters (9 tiles/wave)
__global__ __launch_bounds__(256, 1)
void k_qkv(const __bf16* __restrict__ sf2b, const __bf16* __restrict__ WqB,
           const float* __restrict__ qkvb, __bf16* __restrict__ qkvT) {
  __shared__ __bf16 XL[32][Cc + 8];
  int blk = blockIdx.x, b = blk / 18, n0 = (blk % 18) * 32;
  int tid = threadIdx.x, lane = tid & 31, w = tid >> 5;
  for (int idx = tid; idx < 32 * Cc; idx += 256) {
    int r = idx / Cc, c = idx % Cc;
    XL[r][c] = sf2b[((size_t)b * Mm + n0 + r) * Cc + c];
  }
  __syncthreads();
  int rt = w & 1, colbase = (w >> 1) * 144;
  int roff = (lane < 16) ? 0 : 8, lcol = lane & 15;
  v8f acc[9];
#pragma unroll
  for (int t = 0; t < 9; ++t) acc[t] = v8f_zero();
  for (int kc = 0; kc < Cc; kc += 32) {
    bfrag a = load_a_rm(&XL[rt * 16][kc], Cc + 8, lane);
#pragma unroll
    for (int t = 0; t < 9; ++t) {
      bfrag bb = load_b_cm(WqB + (size_t)(colbase + t * 16) * Cc + kc, Cc, lane);
      WMMA_BF16(acc[t], a, bb);
    }
  }
#pragma unroll
  for (int t = 0; t < 9; ++t) {
    int o = colbase + t * 16 + lcol;
    float bias = qkvb[o];
#pragma unroll
    for (int i = 0; i < 8; ++i)
      qkvT[((size_t)b * TC + n0 + rt * 16 + roff + i) * TC + o] = (__bf16)(acc[t][i] + bias);
  }
}

// ST[m][n] = sum_d q[d][m] k[d][n] * scale   (store row-major so softmax over n is a row softmax)
__global__ __launch_bounds__(256, 1)
void k_attn_s(const __bf16* __restrict__ qkvT, float* __restrict__ STb) {
  __shared__ __bf16 qA[64][HD + 4];
  __shared__ __bf16 qB[64][HD + 4];
  int blk = blockIdx.x;
  int tb = blk % 81, h = (blk / 81) % NH, b = blk / (81 * NH);
  int m0 = (tb / 9) * 64, n0 = (tb % 9) * 64;
  int tid = threadIdx.x, lane = tid & 31, w = tid >> 5;
  int oq = h * 144, ok = h * 144 + 48;
  for (int idx = tid; idx < 64 * HD; idx += 256) {
    int r = idx / HD, d = idx % HD;
    qA[r][d] = qkvT[((size_t)b * TC + m0 + r) * TC + oq + d];
    qB[r][d] = qkvT[((size_t)b * TC + n0 + r) * TC + ok + d];
  }
  __syncthreads();
  int rt = w & 3, ct0 = (w >> 2) * 2;
  int roff = (lane < 16) ? 0 : 8, lcol = lane & 15;
  const float scale = rsqrtf((float)HD);
#pragma unroll
  for (int tt = 0; tt < 2; ++tt) {
    int ct = ct0 + tt;
    v8f acc = v8f_zero();
#pragma unroll
    for (int kc = 0; kc < 64; kc += 32) {
      bfrag a = load_a_rm_g(&qA[rt * 16][0], HD + 4, lane, kc, HD);
      bfrag bb = load_b_cm_g(&qB[ct * 16][0], HD + 4, lane, kc, HD);
      WMMA_BF16(acc, a, bb);
    }
    float* dst = STb + ((size_t)(b * NH + h) * TC + m0 + rt * 16 + roff) * TC + n0 + ct * 16 + lcol;
#pragma unroll
    for (int i = 0; i < 8; ++i) dst[(size_t)i * TC] = acc[i] * scale;
  }
}

__global__ __launch_bounds__(256, 1)
void k_softmax_rows(const float* __restrict__ STb, __bf16* __restrict__ Pb) {
  __shared__ float red[256];
  int row = blockIdx.x, tid = threadIdx.x;
  const float* src = STb + (size_t)row * TC;
  float mx = -1e30f;
  for (int i = tid; i < TC; i += 256) mx = fmaxf(mx, src[i]);
  red[tid] = mx; __syncthreads();
  for (int s = 128; s > 0; s >>= 1) { if (tid < s) red[tid] = fmaxf(red[tid], red[tid + s]); __syncthreads(); }
  mx = red[0]; __syncthreads();
  float sm = 0.f;
  for (int i = tid; i < TC; i += 256) sm += __expf(src[i] - mx);
  red[tid] = sm; __syncthreads();
  for (int s = 128; s > 0; s >>= 1) { if (tid < s) red[tid] += red[tid + s]; __syncthreads(); }
  float inv = 1.f / red[0];
  __bf16* dst = Pb + (size_t)row * TC;
  for (int i = tid; i < TC; i += 256) dst[i] = (__bf16)(__expf(src[i] - mx) * inv);
}

// aoT[m][h*48+d] = sum_n P[m][n] v[d][n]
__global__ __launch_bounds__(128, 1)
void k_attn_o(const __bf16* __restrict__ Pb, const __bf16* __restrict__ qkvT,
              __bf16* __restrict__ aoT) {
  __shared__ __bf16 vL[32][HD + 4];
  int blk = blockIdx.x;
  int mb = blk % 9, h = (blk / 9) % NH, b = blk / (9 * NH);
  int m0 = mb * 64;
  int tid = threadIdx.x, lane = tid & 31, w = tid >> 5;
  int ov = h * 144 + 96;
  int roff = (lane < 16) ? 0 : 8, lcol = lane & 15;
  v8f acc[3];
#pragma unroll
  for (int t = 0; t < 3; ++t) acc[t] = v8f_zero();
  const __bf16* prow = Pb + (size_t)(b * NH + h) * TC * TC;
  for (int kc = 0; kc < TC; kc += 32) {
    for (int idx = tid; idx < 32 * HD; idx += 128) {
      int r = idx / HD, d = idx % HD;
      vL[r][d] = qkvT[((size_t)b * TC + kc + r) * TC + ov + d];
    }
    __syncthreads();
    bfrag a = load_a_rm(prow + (size_t)(m0 + w * 16) * TC + kc, TC, lane);
#pragma unroll
    for (int t = 0; t < 3; ++t) {
      bfrag bb = load_b_rm(&vL[0][t * 16], HD + 4, lane);
      WMMA_BF16(acc[t], a, bb);
    }
    __syncthreads();
  }
#pragma unroll
  for (int t = 0; t < 3; ++t)
#pragma unroll
    for (int i = 0; i < 8; ++i)
      aoT[((size_t)b * TC + m0 + w * 16 + roff + i) * Cc + h * HD + t * 16 + lcol] =
          (__bf16)acc[t][i];
}

// proj: sfrb[n][co] = sum_ci ao[n][ci] Pw[co][ci] + pb[co]
__global__ __launch_bounds__(256, 1)
void k_proj(const __bf16* __restrict__ aoT, const __bf16* __restrict__ WpB,
            const float* __restrict__ pbias, __bf16* __restrict__ sfrb) {
  __shared__ __bf16 XL[64][Cc + 8];
  int blk = blockIdx.x, b = blk / 9, n0 = (blk % 9) * 64;
  int tid = threadIdx.x, lane = tid & 31, w = tid >> 5;
  for (int idx = tid; idx < 64 * Cc; idx += 256) {
    int r = idx / Cc, c = idx % Cc;
    XL[r][c] = aoT[((size_t)b * TC + n0 + r) * Cc + c];
  }
  __syncthreads();
  int rt = w & 3, cbase = (w >> 2) * 96;
  int roff = (lane < 16) ? 0 : 8, lcol = lane & 15;
  v8f acc[6];
#pragma unroll
  for (int t = 0; t < 6; ++t) acc[t] = v8f_zero();
  for (int kc = 0; kc < Cc; kc += 32) {
    bfrag a = load_a_rm(&XL[rt * 16][kc], Cc + 8, lane);
#pragma unroll
    for (int t = 0; t < 6; ++t) {
      bfrag bb = load_b_cm(WpB + (size_t)(cbase + t * 16) * Cc + kc, Cc, lane);
      WMMA_BF16(acc[t], a, bb);
    }
  }
#pragma unroll
  for (int t = 0; t < 6; ++t) {
    int co = cbase + t * 16 + lcol;
    float bias = pbias[co];
#pragma unroll
    for (int i = 0; i < 8; ++i)
      sfrb[((size_t)b * Mm + n0 + rt * 16 + roff + i) * Cc + co] = (__bf16)(acc[t][i] + bias);
  }
}

// redistribute: out_pix[p][c] = sum_k aff[p][k] * refined_neighbor_k[c]
__global__ __launch_bounds__(256, 1)
void k_redist(const float* __restrict__ affG, const __bf16* __restrict__ sfrb,
              float* __restrict__ out) {
  __shared__ __bf16 affL[64][32];
  __shared__ __bf16 sfuL[32][Cc + 8];
  __shared__ float outL[96][66];
  int blk = blockIdx.x, b = blk / Mm, m = blk % Mm;
  int sy = m / WWs, sx = m % WWs;
  int tid = threadIdx.x, lane = tid & 31, w = tid >> 5;
  for (int idx = tid; idx < 64 * 32; idx += 256) {
    int p = idx >> 5, k = idx & 31;
    affL[p][k] = (k < 9) ? (__bf16)affG[(((size_t)b * Mm + m) * 64 + p) * 9 + k] : (__bf16)0.f;
  }
  for (int idx = tid; idx < 32 * Cc; idx += 256) {
    int k = idx / Cc, c = idx % Cc;
    __bf16 v = (__bf16)0.f;
    if (k < 9) {
      int ny = sy + k / 3 - 1, nx = sx + k % 3 - 1;
      if (ny >= 0 && ny < HHs && nx >= 0 && nx < WWs)
        v = sfrb[((size_t)b * Mm + ny * WWs + nx) * Cc + c];
    }
    sfuL[k][c] = v;
  }
  __syncthreads();
  int rt = w & 3, tset = (w >> 2) * 3;
  int roff = (lane < 16) ? 0 : 8, lcol = lane & 15;
  for (int ch = 0; ch < 2; ++ch) {
#pragma unroll
    for (int tt = 0; tt < 3; ++tt) {
      int tl = tset + tt;
      int c0 = ch * 96 + tl * 16;
      v8f acc = v8f_zero();
      bfrag a = load_a_rm(&affL[rt * 16][0], 32, lane);
      bfrag bb = load_b_rm(&sfuL[0][c0], Cc + 8, lane);
      WMMA_BF16(acc, a, bb);
#pragma unroll
      for (int i = 0; i < 8; ++i) outL[tl * 16 + lcol][rt * 16 + roff + i] = acc[i];
    }
    __syncthreads();
    for (int idx = tid; idx < 96 * 64; idx += 256) {
      int lc = idx >> 6, p = idx & 63;
      int c = ch * 96 + lc;
      out[((size_t)(b * Cc + c) * Hh + sy * 8 + (p >> 3)) * Wd + sx * 8 + (p & 7)] = outL[lc][p];
    }
    __syncthreads();
  }
}

// ================= host =================
extern "C" void kernel_launch(void* const* d_in, const int* in_sizes, int n_in,
                              void* d_out, int out_size, void* d_ws, size_t ws_size,
                              hipStream_t stream) {
  const float* x_cur  = (const float*)d_in[0];
  const float* x_ref  = (const float*)d_in[1];
  const float* qkv_w  = (const float*)d_in[2];
  const float* qkv_b  = (const float*)d_in[3];
  const float* proj_w = (const float*)d_in[4];
  const float* proj_b = (const float*)d_in[5];
  float* out = (float*)d_out;
  (void)in_sizes; (void)n_in; (void)out_size; (void)ws_size;

  char* ws = (char*)d_ws;
  size_t off = 0;
  auto alloc = [&](size_t bytes) -> void* {
    void* p = ws + off;
    off += (bytes + 255) & ~(size_t)255;
    return p;
  };
  __bf16* XrT    = (__bf16*)alloc((size_t)Bn * Nn * Cc * 2);
  float*  pnorm  = (float*) alloc((size_t)Bn * Nn * 4);
  __bf16* centT  = (__bf16*)alloc((size_t)Bn * Mm * Cc * 2);
  float*  cnorm  = (float*) alloc((size_t)Bn * Mm * 4);
  __bf16* Abuf   = (__bf16*)alloc((size_t)Bn * Nn * Mm * 2);
  float*  sw     = (float*) alloc((size_t)Bn * Mm * 4);
  float*  wp     = (float*) alloc((size_t)Bn * Mm * Cc * 4);
  __bf16* sfb    = (__bf16*)alloc((size_t)Bn * Mm * Cc * 2);
  float*  affG   = (float*) alloc((size_t)Bn * Mm * Pp * 9 * 4);
  float*  sf2acc = (float*) alloc((size_t)Bn * Mm * Cc * 4);
  float*  asum   = (float*) alloc((size_t)Bn * Mm * 4);
  __bf16* sf2b   = (__bf16*)alloc((size_t)Bn * Mm * Cc * 2);
  __bf16* WqB    = (__bf16*)alloc((size_t)TC * Cc * 2);
  __bf16* WpB    = (__bf16*)alloc((size_t)Cc * Cc * 2);
  __bf16* qkvT   = (__bf16*)alloc((size_t)Bn * TC * TC * 2);
  float*  STb    = (float*) alloc((size_t)Bn * NH * TC * TC * 4);
  __bf16* Pb     = (__bf16*)alloc((size_t)Bn * NH * TC * TC * 2);
  __bf16* aoT    = (__bf16*)alloc((size_t)Bn * TC * Cc * 2);
  __bf16* sfrb   = (__bf16*)alloc((size_t)Bn * Mm * Cc * 2);

  hipMemsetAsync(sw, 0, (size_t)Bn * Mm * 4, stream);
  hipMemsetAsync(wp, 0, (size_t)Bn * Mm * Cc * 4, stream);
  hipMemsetAsync(sf2acc, 0, (size_t)Bn * Mm * Cc * 4, stream);
  hipMemsetAsync(asum, 0, (size_t)Bn * Mm * 4, stream);

  k_pack_xref<<<Bn * (Nn / 64), 256, 0, stream>>>(x_ref, XrT, pnorm);
  k_centroid <<<Bn * Mm, 256, 0, stream>>>(x_ref, centT, cnorm);
  k_pack_w   <<<(TC * Cc + 255) / 256, 256, 0, stream>>>(qkv_w, WqB, TC * Cc);
  k_pack_w   <<<(Cc * Cc + 255) / 256, 256, 0, stream>>>(proj_w, WpB, Cc * Cc);
  k_assign   <<<Bn * (Nn / 32), 256, 0, stream>>>(XrT, centT, pnorm, cnorm, Abuf, sw);
  k_wp       <<<Bn * 9 * 16, 256, 0, stream>>>(Abuf, XrT, wp);
  k_sf_fin   <<<(int)(((size_t)Bn * Mm * Cc + 255) / 256), 256, 0, stream>>>(wp, sw, sfb);
  k_aff      <<<Bn * Mm, 128, 0, stream>>>(x_cur, sfb, affG, sf2acc, asum);
  k_sf2_fin  <<<(int)(((size_t)Bn * Mm * Cc + 255) / 256), 256, 0, stream>>>(sf2acc, asum, sf2b);
  k_qkv      <<<Bn * 18, 256, 0, stream>>>(sf2b, WqB, qkv_b, qkvT);
  k_attn_s   <<<Bn * NH * 81, 256, 0, stream>>>(qkvT, STb);
  k_softmax_rows<<<Bn * NH * TC, 256, 0, stream>>>(STb, Pb);
  k_attn_o   <<<Bn * NH * 9, 128, 0, stream>>>(Pb, qkvT, aoT);
  k_proj     <<<Bn * 9, 256, 0, stream>>>(aoT, WpB, proj_b, sfrb);
  k_redist   <<<Bn * Mm, 256, 0, stream>>>(affG, sfrb, out);
}
